// GCNGNN_6614249636268
// MI455X (gfx1250) — matrix operand order repaired and checked by reading, not compiled, not run on hardware
//
#include <hip/hip_runtime.h>

#define N_NODES 50000
#define N_EDGES 800000
#define DIM     128
#define LAYERS  5
#define LDA     (DIM + 8)   // LDS row pad: 272B stride -> fragment reads spread across banks

typedef __attribute__((ext_vector_type(16))) _Float16 v16h;
typedef __attribute__((ext_vector_type(8)))  _Float16 v8h;
typedef __attribute__((ext_vector_type(8)))  float    v8f;

// ---------------------------------------------------------------- utilities
__global__ __launch_bounds__(256) void zero_kernel(float* __restrict__ p, int n) {
    int i = blockIdx.x * 256 + threadIdx.x;
    if (i < n) p[i] = 0.0f;
}

__global__ __launch_bounds__(256) void degree_kernel(const int* __restrict__ src,
                                                     const int* __restrict__ dst,
                                                     float* __restrict__ deg_out,
                                                     float* __restrict__ deg_in, int E) {
    int e = blockIdx.x * 256 + threadIdx.x;
    if (e < E) {
        atomicAdd(deg_out + src[e], 1.0f);
        atomicAdd(deg_in  + dst[e], 1.0f);
    }
}

__global__ __launch_bounds__(256) void norm_kernel(const float* __restrict__ deg_out,
                                                   const float* __restrict__ deg_in,
                                                   float* __restrict__ norm_src,
                                                   float* __restrict__ norm_dst, int n) {
    int i = blockIdx.x * 256 + threadIdx.x;
    if (i < n) {
        norm_src[i] = rsqrtf(fmaxf(deg_out[i], 1.0f));
        norm_dst[i] = rsqrtf(fmaxf(deg_in[i],  1.0f));
    }
}

// ------------------------------------------------------- W -> f16 B-fragment
// packed f16 B (32x16, 8 VGPRs, wave32), contiguous k order chosen here (A matches):
//   lane L: n = L&15;  element j (0..15): K = kb*32 + (L>>4)*16 + j
// Fragment flat layout: wfrag[ ((l*4 + kb)*8 + nt)*32*16 + lane*16 + j ]
__global__ __launch_bounds__(256) void pack_w_kernel(const float* __restrict__ W,
                                                     _Float16* __restrict__ wfrag, int total) {
    int gid = blockIdx.x * 256 + threadIdx.x;
    if (gid >= total) return;                 // total = LAYERS*DIM*DIM
    int l    = gid >> 14;                     // / 16384
    int r    = gid & 16383;
    int j    = r & 15;
    int lane = (r >> 4) & 31;
    int nt   = (r >> 9) & 7;
    int kb   = (r >> 12) & 3;
    int K = kb * 32 + ((lane >> 4) << 4) + j;
    int n = nt * 16 + (lane & 15);
    wfrag[gid] = (_Float16)W[(size_t)l * 16384 + (size_t)K * DIM + n];
}

// --------------------------------------------------------------- SpMM scatter
// one wave32 per edge; lane covers 4 of the 128 columns (float4 gather, L2-resident)
__global__ __launch_bounds__(256) void scatter_kernel(const int* __restrict__ src,
                                                      const int* __restrict__ dst,
                                                      const float* __restrict__ h,
                                                      const float* __restrict__ norm_src,
                                                      float* __restrict__ agg, int E) {
    int e = blockIdx.x * 8 + (threadIdx.x >> 5);
    if (e >= E) return;
    int lane = threadIdx.x & 31;
    int s = src[e], d = dst[e];
    float ns = norm_src[s];
    const float4 v = *reinterpret_cast<const float4*>(h + (size_t)s * DIM + lane * 4);
    float* ap = agg + (size_t)d * DIM + lane * 4;
    atomicAdd(ap + 0, v.x * ns);
    atomicAdd(ap + 1, v.y * ns);
    atomicAdd(ap + 2, v.z * ns);
    atomicAdd(ap + 3, v.w * ns);
}

// ------------------------------------------- fused scale+convert+GEMM+bias+ReLU
// block = 256 threads = 8 wave32s, computes a 16(M) x 128(N) strip of the output.
// Wave w owns N-tile w; K=128 via 4 chained v_wmma_f32_16x16x32_f16.
__global__ __launch_bounds__(256) void gemm_relu_kernel(const float* __restrict__ agg,
                                                        const float* __restrict__ norm_dst,
                                                        const _Float16* __restrict__ wfragL,
                                                        const float* __restrict__ biasL,
                                                        float* __restrict__ out) {
    __shared__ _Float16 As[16 * LDA];
    const int t    = threadIdx.x;
    const int row0 = blockIdx.x * 16;

    // stage: load agg tile (f32), scale by norm_dst, convert to f16 into LDS
    {
        int r = t >> 4;                 // 0..15 row
        int c = (t & 15) << 3;          // 0..120, 8 elements each
        const float* gp = agg + (size_t)(row0 + r) * DIM + c;
        float s = norm_dst[row0 + r];
        float4 v0 = *reinterpret_cast<const float4*>(gp);
        float4 v1 = *reinterpret_cast<const float4*>(gp + 4);
        v8h hv;
        hv[0] = (_Float16)(v0.x * s); hv[1] = (_Float16)(v0.y * s);
        hv[2] = (_Float16)(v0.z * s); hv[3] = (_Float16)(v0.w * s);
        hv[4] = (_Float16)(v1.x * s); hv[5] = (_Float16)(v1.y * s);
        hv[6] = (_Float16)(v1.z * s); hv[7] = (_Float16)(v1.w * s);
        *reinterpret_cast<v8h*>(&As[r * LDA + c]) = hv;
    }
    __syncthreads();

    const int wave = t >> 5;        // N-tile 0..7
    const int lane = t & 31;
    const int m    = lane & 15;     // A-fragment row
    const int half = lane >> 4;
    const int n    = lane & 15;     // C/D column within tile

    // accumulator initialized with bias (bias depends only on column n)
    float bv = biasL[wave * 16 + n];
    v8f acc;
#pragma unroll
    for (int r = 0; r < 8; ++r) acc[r] = bv;

    const v16h* bp = reinterpret_cast<const v16h*>(wfragL);
#pragma unroll
    for (int kb = 0; kb < 4; ++kb) {
        const int k0 = kb * 32;
        // A-fragment (matches packed B): element j -> K=k0+half*16+j
        const v8h alo = *reinterpret_cast<const v8h*>(&As[m * LDA + k0 + half * 16]);
        const v8h ahi = *reinterpret_cast<const v8h*>(&As[m * LDA + k0 + half * 16 + 8]);
        v16h a;
#pragma unroll
        for (int j = 0; j < 8; ++j) { a[j] = alo[j]; a[8 + j] = ahi[j]; }
        // B-fragment: prepacked, one contiguous 32B vector per lane
        v16h b = bp[(size_t)(kb * 8 + wave) * 32 + lane];
        acc = __builtin_amdgcn_wmma_f32_16x16x32_f16(false, a, false, b,
                                                     (short)0, acc, false, false);
    }

    // D layout: lane L holds (m = r + 8*(L>>4), n = L&15) in vgpr r
    float* op = out + (size_t)row0 * DIM + wave * 16 + n;
#pragma unroll
    for (int r = 0; r < 8; ++r) {
        int mo = r + half * 8;
        op[(size_t)mo * DIM] = fmaxf(acc[r], 0.0f);
    }
}

// ------------------------------------------------------------------- launcher
extern "C" void kernel_launch(void* const* d_in, const int* in_sizes, int n_in,
                              void* d_out, int out_size, void* d_ws, size_t ws_size,
                              hipStream_t stream) {
    const float* feat = (const float*)d_in[0];
    const int*   src  = (const int*)d_in[1];
    const int*   dst  = (const int*)d_in[2];
    const float* W    = (const float*)d_in[3];
    const float* bias = (const float*)d_in[4];
    float*       out  = (float*)d_out;

    const int N = N_NODES, E = N_EDGES;

    char* ws = (char*)d_ws;
    auto alloc = [&](size_t bytes) -> char* {
        char* p = ws;
        ws += (bytes + 255) & ~size_t(255);
        return p;
    };
    float*    deg      = (float*)alloc((size_t)2 * N * sizeof(float));
    float*    deg_out  = deg;
    float*    deg_in   = deg + N;
    float*    norm_src = (float*)alloc((size_t)N * sizeof(float));
    float*    norm_dst = (float*)alloc((size_t)N * sizeof(float));
    float*    agg      = (float*)alloc((size_t)N * DIM * sizeof(float));
    float*    hbuf     = (float*)alloc((size_t)N * DIM * sizeof(float));
    _Float16* wfrag    = (_Float16*)alloc((size_t)LAYERS * DIM * DIM * sizeof(_Float16));

    zero_kernel<<<(2 * N + 255) / 256, 256, 0, stream>>>(deg, 2 * N);
    degree_kernel<<<(E + 255) / 256, 256, 0, stream>>>(src, dst, deg_out, deg_in, E);
    norm_kernel<<<(N + 255) / 256, 256, 0, stream>>>(deg_out, deg_in, norm_src, norm_dst, N);
    pack_w_kernel<<<(LAYERS * DIM * DIM + 255) / 256, 256, 0, stream>>>(
        W, wfrag, LAYERS * DIM * DIM);

    const float* hin = feat;
    for (int l = 0; l < LAYERS; ++l) {
        zero_kernel<<<(N * DIM + 255) / 256, 256, 0, stream>>>(agg, N * DIM);
        scatter_kernel<<<(E + 7) / 8, 256, 0, stream>>>(src, dst, hin, norm_src, agg, E);
        float* hout = (l == LAYERS - 1) ? out : hbuf;
        gemm_relu_kernel<<<N / 16, 256, 0, stream>>>(
            agg, norm_dst, wfrag + (size_t)l * DIM * DIM, bias + (size_t)l * DIM, hout);
        hin = hbuf;
    }
}
